// EagerMultiheadAttention_27616639713994
// MI455X (gfx1250) — matrix operand order repaired and checked by reading, not compiled, not run on hardware
//
#include <hip/hip_runtime.h>
#include <hip/hip_bf16.h>
#include <cstddef>

// Problem constants (from reference)
#define T_DIM 2048
#define S_DIM 2048
#define B_DIM 2
#define E_DIM 1024
#define H_DIM 16
#define HD    64

#define PADB  2056   // bf16 LDS row stride for P planes (pad -> conflict-reduced frag loads)

typedef __attribute__((ext_vector_type(16))) __bf16 v16bf;
typedef __attribute__((ext_vector_type(8)))  __bf16 v8bf;
typedef __attribute__((ext_vector_type(8)))  float  v8f;

// ---------------------------------------------------------------------------
// WMMA: D = A(16x32 bf16) * B(32x16 bf16) + C(16x16 f32)
// ---------------------------------------------------------------------------
__device__ __forceinline__ v8f wmma_bf16(v16bf a, v16bf b, v8f c) {
    return __builtin_amdgcn_wmma_f32_16x16x32_bf16(
        false, a, false, b, (short)0, c, false, false);
}

// bf16x2 (hi/lo) split multiply-accumulate: C += (Ah+Al)*(Bh+Bl), dropping lo*lo
__device__ __forceinline__ v8f mma3(v16bf ah, v16bf al, v16bf bh, v16bf bl, v8f c) {
    c = wmma_bf16(ah, bh, c);
    c = wmma_bf16(ah, bl, c);
    c = wmma_bf16(al, bh, c);
    return c;
}

__device__ __forceinline__ void split2(float x, __bf16& h, __bf16& l) {
    h = (__bf16)x;
    l = (__bf16)(x - (float)h);
}

__device__ __forceinline__ v16bf cat8(v8bf x, v8bf y) {
    return __builtin_shufflevector(x, y, 0,1,2,3,4,5,6,7,8,9,10,11,12,13,14,15);
}

// ---------------------------------------------------------------------------
// Fragment loaders (ISA 7.12.2, wave32). Pointers may be global or LDS.
// A 16x32 (MxK): lane m=l&15; kk=(l&16)?8:0; elems 0..7 -> kk..kk+7, 8..15 -> kk+16..kk+23
// ---------------------------------------------------------------------------
__device__ __forceinline__ v16bf load_af(const __bf16* rowA, int kbase, int lane) {
    const int kk = kbase + ((lane & 16) ? 8 : 0);
    v8bf x0 = *(const v8bf*)(rowA + kk);
    v8bf x1 = *(const v8bf*)(rowA + kk + 16);
    return cat8(x0, x1);
}

// B 32x16 (KxN): lane n=l&15; kb=(l&16)?8:0; elems 0..7 -> kb..kb+7, 8..15 -> kb+16..kb+23
__device__ __forceinline__ v16bf load_bf(const __bf16* rowB, int kbase, int lane) {
    const int kb = kbase + ((lane & 16) ? 8 : 0);
    v8bf x0 = *(const v8bf*)(rowB + kb);
    v8bf x1 = *(const v8bf*)(rowB + kb + 16);
    return cat8(x0, x1);
}

// ---------------------------------------------------------------------------
// Kernel 0: one-time fp32 -> (bf16 hi, bf16 lo) plane split
// ---------------------------------------------------------------------------
__global__ __launch_bounds__(256)
void split_kernel(const float* __restrict__ in, __bf16* __restrict__ hi,
                  __bf16* __restrict__ lo, int n) {
    for (int i = blockIdx.x * 256 + threadIdx.x; i < n; i += gridDim.x * 256) {
        float x = in[i];
        __bf16 h, l;
        split2(x, h, l);
        hi[i] = h;
        lo[i] = l;
    }
}

// ---------------------------------------------------------------------------
// GEMM: Out(4096,1024) = A(4096x1024) @ W^T(1024x1024) + bias, all bf16x2 planes.
// 8 waves/WG; WG tile 128x64; wave tile 32x32.
// MODE 0: f32 out (final projection)
// MODE 1: split bf16 planes out (Q, K)
// MODE 2: split bf16 planes, transposed to [b*E+n][t]  (V for the P@V GEMM)
// ---------------------------------------------------------------------------
template <int MODE>
__global__ __launch_bounds__(256)
void proj_gemm_kernel(const __bf16* __restrict__ Ahi, const __bf16* __restrict__ Alo,
                      const __bf16* __restrict__ Whi, const __bf16* __restrict__ Wlo,
                      const float* __restrict__ bias,
                      float* __restrict__ OutF,
                      __bf16* __restrict__ OutHi, __bf16* __restrict__ OutLo) {
    const int lane = threadIdx.x & 31;
    const int wave = threadIdx.x >> 5;
    const int m0 = blockIdx.x * 128 + (wave & 3) * 32;
    const int n0 = blockIdx.y * 64  + (wave >> 2) * 32;

    const size_t aoff0 = (size_t)(m0 +      (lane & 15)) * E_DIM;
    const size_t aoff1 = (size_t)(m0 + 16 + (lane & 15)) * E_DIM;
    const size_t boff0 = (size_t)(n0 +      (lane & 15)) * E_DIM;
    const size_t boff1 = (size_t)(n0 + 16 + (lane & 15)) * E_DIM;

    v8f acc[2][2] = {};
    for (int k = 0; k < E_DIM; k += 32) {
        if (k + 64 < E_DIM) __builtin_prefetch(Ahi + aoff0 + k + 64, 0, 0);
        v16bf a0h = load_af(Ahi + aoff0, k, lane);
        v16bf a0l = load_af(Alo + aoff0, k, lane);
        v16bf a1h = load_af(Ahi + aoff1, k, lane);
        v16bf a1l = load_af(Alo + aoff1, k, lane);
        v16bf b0h = load_bf(Whi + boff0, k, lane);
        v16bf b0l = load_bf(Wlo + boff0, k, lane);
        v16bf b1h = load_bf(Whi + boff1, k, lane);
        v16bf b1l = load_bf(Wlo + boff1, k, lane);
        acc[0][0] = mma3(a0h, a0l, b0h, b0l, acc[0][0]);
        acc[0][1] = mma3(a0h, a0l, b1h, b1l, acc[0][1]);
        acc[1][0] = mma3(a1h, a1l, b0h, b0l, acc[1][0]);
        acc[1][1] = mma3(a1h, a1l, b1h, b1l, acc[1][1]);
    }

    // C/D layout: row m = mi*16 + (lane>>4)*8 + r, col n = ni*16 + (lane&15)
#pragma unroll
    for (int ni = 0; ni < 2; ++ni) {
        const int n = n0 + ni * 16 + (lane & 15);
        const float bv = bias[n];
#pragma unroll
        for (int mi = 0; mi < 2; ++mi) {
#pragma unroll
            for (int r = 0; r < 8; ++r) {
                const int m = m0 + mi * 16 + ((lane >> 4) * 8) + r;
                const float v = acc[mi][ni][r] + bv;
                if (MODE == 0) {
                    OutF[(size_t)m * E_DIM + n] = v;
                } else if (MODE == 1) {
                    __bf16 h, l;
                    split2(v, h, l);
                    OutHi[(size_t)m * E_DIM + n] = h;
                    OutLo[(size_t)m * E_DIM + n] = l;
                } else {                       // MODE 2: V transposed (B_DIM == 2)
                    const int t  = m >> 1;
                    const int bb = m & 1;
                    const size_t ro = (size_t)(bb * E_DIM + n) * S_DIM + t;
                    __bf16 h, l;
                    split2(v, h, l);
                    OutHi[ro] = h;
                    OutLo[ro] = l;
                }
            }
        }
    }
}

// ---------------------------------------------------------------------------
// Attention core: one WG per (b, 16 query rows); loops all 16 heads, so the
// avg_w accumulation is WG-private (no atomics). Score strip lives in LDS as
// split bf16 planes (hi+lo ~ fp32) -> P@V A-fragments are raw ds_load_b128.
// ---------------------------------------------------------------------------
__global__ __launch_bounds__(256)
void attn_kernel(const __bf16* __restrict__ Qhi, const __bf16* __restrict__ Qlo,
                 const __bf16* __restrict__ Khi, const __bf16* __restrict__ Klo,
                 const __bf16* __restrict__ Vthi, const __bf16* __restrict__ Vtlo,
                 __bf16* __restrict__ CtxHi, __bf16* __restrict__ CtxLo,
                 float* __restrict__ avg) {
    extern __shared__ __align__(16) char smemraw[];
    __bf16* Phi = (__bf16*)smemraw;                 // 16 * PADB
    __bf16* Plo = Phi + 16 * PADB;                  // 16 * PADB
    float* red     = (float*)(Plo + 16 * PADB);     // 256
    float* rowstat = red + 256;                     // 16
    float* Obuf    = rowstat + 16;                  // 8 * 256

    const int t0   = blockIdx.x * 16;
    const int b    = blockIdx.y;
    const int tid  = threadIdx.x;
    const int lane = tid & 31;
    const int wave = tid >> 5;
    const float scaling = 0.125f;                   // hd^-0.5
    const float invH = 1.0f / (float)H_DIM;

    for (int h = 0; h < H_DIM; ++h) {
        // ---- q fragments for this head (A 16x64), built once per head ----
        const size_t qoff = ((size_t)(t0 + (lane & 15)) * B_DIM + b) * E_DIM + h * HD;
        v16bf qa0h = load_af(Qhi + qoff, 0,  lane);
        v16bf qa0l = load_af(Qlo + qoff, 0,  lane);
        v16bf qa1h = load_af(Qhi + qoff, 32, lane);
        v16bf qa1l = load_af(Qlo + qoff, 32, lane);

        // ---- scores: each wave covers s = wave*16, stride 128 ----
        for (int st = wave * 16; st < S_DIM; st += 128) {
            const size_t koff = ((size_t)(st + (lane & 15)) * B_DIM + b) * E_DIM + h * HD;
            v16bf kb0h = load_bf(Khi + koff, 0,  lane);
            v16bf kb0l = load_bf(Klo + koff, 0,  lane);
            v16bf kb1h = load_bf(Khi + koff, 32, lane);
            v16bf kb1l = load_bf(Klo + koff, 32, lane);
            v8f c = {};
            c = mma3(qa0h, qa0l, kb0h, kb0l, c);
            c = mma3(qa1h, qa1l, kb1h, kb1l, c);
#pragma unroll
            for (int r = 0; r < 8; ++r) {
                const int m = ((lane >> 4) * 8) + r;
                __bf16 ph, pl;
                split2(c[r] * scaling, ph, pl);
                Phi[m * PADB + st + (lane & 15)] = ph;
                Plo[m * PADB + st + (lane & 15)] = pl;
            }
        }
        __syncthreads();

        // ---- softmax: 16 threads per row, 128 elems each ----
        const int row  = tid & 15;
        const int part = tid >> 4;
        __bf16* phr = Phi + row * PADB + part * 128;
        __bf16* plr = Plo + row * PADB + part * 128;

        float mx = -1e30f;
        for (int i = 0; i < 128; ++i)
            mx = fmaxf(mx, (float)phr[i] + (float)plr[i]);
        red[row * 16 + part] = mx;
        __syncthreads();
        if (tid < 16) {
            float m2 = red[tid * 16];
            for (int i = 1; i < 16; ++i) m2 = fmaxf(m2, red[tid * 16 + i]);
            rowstat[tid] = m2;
        }
        __syncthreads();
        const float rmax = rowstat[row];
        float lsum = 0.0f;
        for (int i = 0; i < 128; ++i) {
            float x = (float)phr[i] + (float)plr[i];
            float p = __expf(x - rmax);
            lsum += p;
            __bf16 hh, ll;
            split2(p, hh, ll);
            phr[i] = hh;
            plr[i] = ll;
        }
        __syncthreads();                             // rowstat reuse hazard
        red[row * 16 + part] = lsum;
        __syncthreads();
        if (tid < 16) {
            float s2 = 0.0f;
            for (int i = 0; i < 16; ++i) s2 += red[tid * 16 + i];
            rowstat[tid] = 1.0f / s2;
        }
        __syncthreads();
        const float rinv = rowstat[row];

        // normalize in LDS + accumulate averaged weights (no atomics needed)
        float* avgrow = avg + ((size_t)b * T_DIM + t0 + row) * S_DIM + part * 128;
        for (int i = 0; i < 128; ++i) {
            float p  = (float)phr[i] + (float)plr[i];
            float pn = p * rinv;
            __bf16 hh, ll;
            split2(pn, hh, ll);
            phr[i] = hh;
            plr[i] = ll;
            float a = pn * invH;
            if (h == 0) avgrow[i] = a;
            else        avgrow[i] += a;
        }
        __syncthreads();

        // ---- O = P(16x2048) @ V(2048x64): wave -> dtile = wave&3, s-half = wave>>2
        {
            const int d0   = (wave & 3) * 16;
            const int sbeg = (wave >> 2) * 1024;
            const __bf16* prowH = Phi + (lane & 15) * PADB;
            const __bf16* prowL = Plo + (lane & 15) * PADB;
            const size_t voff = (size_t)(b * E_DIM + h * HD + d0 + (lane & 15)) * S_DIM;
            v8f c = {};
            for (int s = sbeg; s < sbeg + 1024; s += 32) {
                v16bf pah = load_af(prowH, s, lane);
                v16bf pal = load_af(prowL, s, lane);
                v16bf vbh = load_bf(Vthi + voff, s, lane);
                v16bf vbl = load_bf(Vtlo + voff, s, lane);
                c = mma3(pah, pal, vbh, vbl, c);
            }
#pragma unroll
            for (int r = 0; r < 8; ++r) {
                const int m = (lane >> 4) * 8 + r;
                Obuf[wave * 256 + m * 16 + (lane & 15)] = c[r];
            }
        }
        __syncthreads();

        // combine the two s-halves and store split context
        for (int idx = tid; idx < 16 * 64; idx += 256) {
            const int m  = idx >> 6;
            const int d  = idx & 63;
            const int w0 = d >> 4;
            float o = Obuf[w0 * 256 + m * 16 + (d & 15)]
                    + Obuf[(w0 + 4) * 256 + m * 16 + (d & 15)];
            const size_t co = ((size_t)(t0 + m) * B_DIM + b) * E_DIM + h * HD + d;
            __bf16 hh, ll;
            split2(o, hh, ll);
            CtxHi[co] = hh;
            CtxLo[co] = ll;
        }
        __syncthreads();
    }
}

// ---------------------------------------------------------------------------
// Host-side launcher
// ---------------------------------------------------------------------------
extern "C" void kernel_launch(void* const* d_in, const int* in_sizes, int n_in,
                              void* d_out, int out_size, void* d_ws, size_t ws_size,
                              hipStream_t stream) {
    (void)in_sizes; (void)n_in; (void)out_size; (void)ws_size;

    const float* query = (const float*)d_in[0];
    const float* key   = (const float*)d_in[1];
    const float* value = (const float*)d_in[2];
    const float* Wq = (const float*)d_in[3];
    const float* bq = (const float*)d_in[4];
    const float* Wk = (const float*)d_in[5];
    const float* bk = (const float*)d_in[6];
    const float* Wv = (const float*)d_in[7];
    const float* bv = (const float*)d_in[8];
    const float* Wo = (const float*)d_in[9];
    const float* bo = (const float*)d_in[10];

    float* out = (float*)d_out;                                   // (T,B,E) f32
    float* avg = out + (size_t)T_DIM * B_DIM * E_DIM;             // (B,T,S) f32

    const size_t SZX = (size_t)T_DIM * B_DIM * E_DIM;             // 4096*1024
    const size_t SZW = (size_t)E_DIM * E_DIM;                     // 1024*1024

    __bf16* p = (__bf16*)d_ws;
    __bf16 *Xq_h = p, *Xq_l = p + SZX; p += 2 * SZX;
    __bf16 *Xk_h = p, *Xk_l = p + SZX; p += 2 * SZX;
    __bf16 *Xv_h = p, *Xv_l = p + SZX; p += 2 * SZX;
    __bf16 *Wq_h = p, *Wq_l = p + SZW; p += 2 * SZW;
    __bf16 *Wk_h = p, *Wk_l = p + SZW; p += 2 * SZW;
    __bf16 *Wv_h = p, *Wv_l = p + SZW; p += 2 * SZW;
    __bf16 *Wo_h = p, *Wo_l = p + SZW; p += 2 * SZW;
    __bf16 *Q_h  = p, *Q_l  = p + SZX; p += 2 * SZX;
    __bf16 *K_h  = p, *K_l  = p + SZX; p += 2 * SZX;
    __bf16 *Vt_h = p, *Vt_l = p + SZX; p += 2 * SZX;
    __bf16 *C_h  = p, *C_l  = p + SZX; p += 2 * SZX;

    // 0) one-time fp32 -> bf16 hi/lo plane splits
    split_kernel<<<1024, 256, 0, stream>>>(query, Xq_h, Xq_l, (int)SZX);
    split_kernel<<<1024, 256, 0, stream>>>(key,   Xk_h, Xk_l, (int)SZX);
    split_kernel<<<1024, 256, 0, stream>>>(value, Xv_h, Xv_l, (int)SZX);
    split_kernel<<<512,  256, 0, stream>>>(Wq, Wq_h, Wq_l, (int)SZW);
    split_kernel<<<512,  256, 0, stream>>>(Wk, Wk_h, Wk_l, (int)SZW);
    split_kernel<<<512,  256, 0, stream>>>(Wv, Wv_h, Wv_l, (int)SZW);
    split_kernel<<<512,  256, 0, stream>>>(Wo, Wo_h, Wo_l, (int)SZW);

    // 1) projections (WMMA, conversion-free inner loops)
    dim3 gemm_grid(T_DIM * B_DIM / 128, E_DIM / 64, 1);
    proj_gemm_kernel<1><<<gemm_grid, 256, 0, stream>>>(Xq_h, Xq_l, Wq_h, Wq_l, bq,
                                                       nullptr, Q_h, Q_l);
    proj_gemm_kernel<1><<<gemm_grid, 256, 0, stream>>>(Xk_h, Xk_l, Wk_h, Wk_l, bk,
                                                       nullptr, K_h, K_l);
    proj_gemm_kernel<2><<<gemm_grid, 256, 0, stream>>>(Xv_h, Xv_l, Wv_h, Wv_l, bv,
                                                       nullptr, Vt_h, Vt_l);

    // 2) attention core (scores + softmax + avg_w + P@V)
    const size_t shmem = (size_t)(2 * 16 * PADB) * sizeof(__bf16)
                       + (size_t)(256 + 16 + 8 * 256) * sizeof(float);
    attn_kernel<<<dim3(T_DIM / 16, B_DIM, 1), 256, shmem, stream>>>(
        Q_h, Q_l, K_h, K_l, Vt_h, Vt_l, C_h, C_l, avg);

    // 3) output projection -> fp32 result
    proj_gemm_kernel<0><<<gemm_grid, 256, 0, stream>>>(C_h, C_l, Wo_h, Wo_l, bo,
                                                       out, nullptr, nullptr);
}
